// GATDecoder_87522843561616
// MI455X (gfx1250) — matrix operand order repaired and checked by reading, not compiled, not run on hardware
//
#include <hip/hip_runtime.h>

// ---- CDNA5 WMMA vector types ----
typedef __attribute__((ext_vector_type(16))) __bf16 v16bf;
typedef __attribute__((ext_vector_type(8)))  __bf16 v8bf;
typedef __attribute__((ext_vector_type(8)))  float  v8f;

#define NNODES 128
#define NPAIRS 8128           // 128*127/2
#define FD 256                // feature dim D
#define NLAYER 3
#define BN 4096               // batch / graph node count
#define EE 131072             // edges
#define ETOT (EE + BN)        // edges + self loops = 135168
#define FCN (2 * NPAIRS)      // 16256

// round-to-nearest-even f32 -> bf16 bits
__device__ __forceinline__ unsigned short f2bf(float f) {
    unsigned int u = __float_as_uint(f);
    unsigned int r = u + 0x7FFFu + ((u >> 16) & 1u);
    return (unsigned short)(r >> 16);
}

__device__ __forceinline__ void atomicMaxF(float* addr, float val) {
    int* ia = (int*)addr;
    int old = *ia;
    while (__int_as_float(old) < val) {
        int prev = atomicCAS(ia, old, __float_as_int(val));
        if (prev == old) break;
        old = prev;
    }
}

__device__ __forceinline__ void edge_sd(const int* __restrict__ ei, int e, int& s, int& d) {
    if (e < EE) { s = ei[e]; d = ei[EE + e]; }
    else        { s = e - EE; d = e - EE; }   // self loop
}

// ---------------- cast / transpose kernels ----------------
__global__ void k_cast_x(const float* __restrict__ x, unsigned short* __restrict__ xbf, int n) {
    int i = blockIdx.x * blockDim.x + threadIdx.x;
    if (i < n) xbf[i] = f2bf(x[i]);
}

// Wt[l][n][k] = bf16(W[l][k][n])   (L x D x D)
__global__ void k_cast_wt(const float* __restrict__ W, unsigned short* __restrict__ wt) {
    int i = blockIdx.x * blockDim.x + threadIdx.x;
    if (i >= NLAYER * FD * FD) return;
    int l = i / (FD * FD);
    int r = i - l * FD * FD;
    int n = r / FD, k = r - n * FD;
    wt[i] = f2bf(W[l * FD * FD + k * FD + n]);
}

// fcWt[n][k] = bf16(fcW[k][n])   (FCN x D)
__global__ void k_cast_fcwt(const float* __restrict__ W, unsigned short* __restrict__ wt) {
    int i = blockIdx.x * blockDim.x + threadIdx.x;
    if (i >= FCN * FD) return;
    int n = i / FD, k = i - n * FD;
    wt[i] = f2bf(W[k * FCN + n]);
}

// ---------------- WMMA GEMM: C[M x N] = A[M x K](bf16,row) * Bt[N x K](bf16,row)^T ----
// one wave per 16x16 output tile, 4 waves / block, K multiple of 32
__global__ void k_wmma_gemm(const unsigned short* __restrict__ A,
                            const unsigned short* __restrict__ Bt,
                            const float* __restrict__ bias,
                            float* __restrict__ C,
                            int tilesN, int K, int ldc) {
    int tile = blockIdx.x * 4 + (threadIdx.x >> 5);
    int tm = tile / tilesN;
    int tn = tile - tm * tilesN;
    int lane = threadIdx.x & 31;
    int lr   = lane & 15;
    int hi   = lane >> 4;          // 0 | 1 : which half-wave

    // A fragment: lane(m=lr) holds 16 contiguous K values = hi*16..hi*16+15
    const unsigned short* ap = A + (size_t)(tm * 16 + lr) * K + hi * 16;
    // B fragment: lane(n=lr) holds 16 contiguous K values, second half-wave K+16
    const unsigned short* bp = Bt + (size_t)(tn * 16 + lr) * K + hi * 16;

    v8f acc = {};
    for (int kb = 0; kb < K; kb += 32) {
        v8bf alo = *(const v8bf*)(ap + kb);
        v8bf ahi = *(const v8bf*)(ap + kb + 8);
        v16bf av = __builtin_shufflevector(alo, ahi,
                     0,1,2,3,4,5,6,7,8,9,10,11,12,13,14,15);
        v16bf bv = *(const v16bf*)(bp + kb);
        __builtin_prefetch(bp + kb + 64, 0, 3);   // global_prefetch_b8 (L2 streaming)
        acc = __builtin_amdgcn_wmma_f32_16x16x32_bf16(
                  false, av, false, bv, (short)0, acc, false, false);
    }

    // C/D layout: VGPR r -> (M = r + 8*hi, N = lr)
    int col  = tn * 16 + lr;
    int row0 = tm * 16 + (hi << 3);
    float bb = bias ? bias[col] : 0.0f;
#pragma unroll
    for (int r = 0; r < 8; ++r)
        C[(size_t)(row0 + r) * ldc + col] = acc[r] + bb;
}

// ---------------- GAT attention pieces ----------------
__global__ void k_attn_scores(const float* __restrict__ h, const float* __restrict__ asrc,
                              const float* __restrict__ adst,
                              float* __restrict__ as_, float* __restrict__ ad_) {
    __shared__ float s1[256], s2[256];
    int i = blockIdx.x, t = threadIdx.x;
    float hv = h[(size_t)i * FD + t];
    s1[t] = hv * asrc[t];
    s2[t] = hv * adst[t];
    __syncthreads();
    for (int off = 128; off > 0; off >>= 1) {
        if (t < off) { s1[t] += s1[t + off]; s2[t] += s2[t + off]; }
        __syncthreads();
    }
    if (t == 0) { as_[i] = s1[0]; ad_[i] = s2[0]; }
}

__global__ void k_init(float* __restrict__ xnext, float* __restrict__ smax,
                       float* __restrict__ ssum, int n) {
    int i = blockIdx.x * blockDim.x + threadIdx.x;
    if (i < n)  xnext[i] = 0.0f;
    if (i < BN) { smax[i] = -3.0e38f; ssum[i] = 0.0f; }
}

__global__ void k_edge_max(const int* __restrict__ ei, const float* __restrict__ as_,
                           const float* __restrict__ ad_, float* __restrict__ smax) {
    int e = blockIdx.x * blockDim.x + threadIdx.x;
    if (e >= ETOT) return;
    int s, d; edge_sd(ei, e, s, d);
    float v = as_[s] + ad_[d];
    v = (v >= 0.0f) ? v : 0.2f * v;
    atomicMaxF(&smax[d], v);
}

__global__ void k_edge_exp(const int* __restrict__ ei, const float* __restrict__ as_,
                           const float* __restrict__ ad_, const float* __restrict__ smax,
                           float* __restrict__ ssum, float* __restrict__ anum) {
    int e = blockIdx.x * blockDim.x + threadIdx.x;
    if (e >= ETOT) return;
    int s, d; edge_sd(ei, e, s, d);
    float v = as_[s] + ad_[d];
    v = (v >= 0.0f) ? v : 0.2f * v;
    float ex = __expf(v - smax[d]);
    anum[e] = ex;
    atomicAdd(&ssum[d], ex);
}

__global__ void k_edge_agg(const int* __restrict__ ei, const float* __restrict__ h,
                           const float* __restrict__ anum, const float* __restrict__ ssum,
                           float* __restrict__ xnext) {
    int e = blockIdx.x;          // one block per edge
    int f = threadIdx.x;         // feature
    int s, d; edge_sd(ei, e, s, d);
    float alpha = anum[e] / ssum[d];
    atomicAdd(&xnext[(size_t)d * FD + f], alpha * h[(size_t)s * FD + f]);
}

__global__ void k_finalize(const float* __restrict__ xnext, const float* __restrict__ gbias,
                           unsigned short* __restrict__ xbf, int n) {
    int i = blockIdx.x * blockDim.x + threadIdx.x;
    if (i >= n) return;
    int f = i & (FD - 1);
    xbf[i] = f2bf(xnext[i] + gbias[f]);
}

// ---------------- Gumbel straight-through -> adjacency flags ----------------
__global__ void k_adj(const float* __restrict__ logits, const float* __restrict__ gu,
                      unsigned char* __restrict__ adjb, int n) {
    int idx = blockIdx.x * blockDim.x + threadIdx.x;
    if (idx >= n) return;
    int node = idx / NPAIRS;
    int p    = idx - node * NPAIRS;
    float l0 = logits[(size_t)node * FCN + 2 * p];
    float l1 = logits[(size_t)node * FCN + 2 * p + 1];
    float u0 = gu[(size_t)idx * 2];
    float u1 = gu[(size_t)idx * 2 + 1];
    float g0 = -__logf(-__logf(u0 + 1e-10f) + 1e-10f);
    float g1 = -__logf(-__logf(u1 + 1e-10f) + 1e-10f);
    // forward value of straight-through = hard one-hot; adj = channel 0 wins
    adjb[idx] = ((l0 + g0) >= (l1 + g1)) ? 1 : 0;
}

__global__ void k_build_A(const unsigned char* __restrict__ adjb, float* __restrict__ out) {
    int idx = blockIdx.x * blockDim.x + threadIdx.x;   // BN * 128 * 128 threads
    int n  = idx >> 14;
    int ij = idx & 16383;
    int i  = ij >> 7;
    int j  = ij & 127;
    float v = 0.0f;
    if (i != j) {
        int a = (i < j) ? i : j;
        int b = (i < j) ? j : i;
        int p = (a * (2 * NNODES - a - 1)) / 2 + (b - a - 1);   // triu_indices(k=1) order
        v = adjb[(size_t)n * NPAIRS + p] ? 1.0f : 0.0f;
    }
    out[idx] = v;
}

// ---------------- launcher ----------------
extern "C" void kernel_launch(void* const* d_in, const int* in_sizes, int n_in,
                              void* d_out, int out_size, void* d_ws, size_t ws_size,
                              hipStream_t stream) {
    const float* x    = (const float*)d_in[0];
    const int*   ei   = (const int*)d_in[1];
    const float* gu   = (const float*)d_in[2];
    const float* gatW = (const float*)d_in[3];
    const float* asrc = (const float*)d_in[4];
    const float* adst = (const float*)d_in[5];
    const float* gatb = (const float*)d_in[6];
    const float* fcW  = (const float*)d_in[7];
    const float* fcb  = (const float*)d_in[8];

    char* ws = (char*)d_ws;
    unsigned short* xbf  = (unsigned short*)(ws + 0);          // 2,097,152 B
    unsigned short* wt   = (unsigned short*)(ws + 2097152);    //   393,216 B
    unsigned short* fcwt = (unsigned short*)(ws + 2490368);    // 8,323,072 B
    float* h     = (float*)(ws + 10813440);                    // 4,194,304 B
    float* xnext = (float*)(ws + 15007744);                    // 4,194,304 B
    float* as_   = (float*)(ws + 19202048);                    //    16,384 B
    float* ad_   = (float*)(ws + 19218432);                    //    16,384 B
    float* smax  = (float*)(ws + 19234816);                    //    16,384 B
    float* ssum  = (float*)(ws + 19251200);                    //    16,384 B
    float* anum  = (float*)(ws + 19267584);                    //   540,672 B
    unsigned char* adjb = (unsigned char*)(ws + 19808256);     // 33,292,288 B
    float* logits = (float*)d_out;   // 4096 x 16256 fits inside 4096 x 16384 d_out

    k_cast_x   <<<(BN * FD + 255) / 256, 256, 0, stream>>>(x, xbf, BN * FD);
    k_cast_wt  <<<(NLAYER * FD * FD + 255) / 256, 256, 0, stream>>>(gatW, wt);
    k_cast_fcwt<<<(FCN * FD + 255) / 256, 256, 0, stream>>>(fcW, fcwt);

    for (int l = 0; l < NLAYER; ++l) {
        // h = x @ W[l]  (bf16 WMMA, f32 accumulate)
        k_wmma_gemm<<<(BN / 16) * (FD / 16) / 4, 128, 0, stream>>>(
            xbf, wt + (size_t)l * FD * FD, nullptr, h, FD / 16, FD, FD);
        k_attn_scores<<<BN, 256, 0, stream>>>(h, asrc + l * FD, adst + l * FD, as_, ad_);
        k_init<<<(BN * FD + 255) / 256, 256, 0, stream>>>(xnext, smax, ssum, BN * FD);
        k_edge_max<<<(ETOT + 255) / 256, 256, 0, stream>>>(ei, as_, ad_, smax);
        k_edge_exp<<<(ETOT + 255) / 256, 256, 0, stream>>>(ei, as_, ad_, smax, ssum, anum);
        k_edge_agg<<<ETOT, 256, 0, stream>>>(ei, h, anum, ssum, xnext);
        k_finalize<<<(BN * FD + 255) / 256, 256, 0, stream>>>(xnext, gatb + l * FD, xbf, BN * FD);
    }

    // logits = x @ fc_W + fc_b  (into d_out scratch)
    k_wmma_gemm<<<(BN / 16) * (FCN / 16) / 4, 128, 0, stream>>>(
        xbf, fcwt, fcb, logits, FCN / 16, FD, FCN);

    k_adj<<<(BN * NPAIRS + 255) / 256, 256, 0, stream>>>(logits, gu, adjb, BN * NPAIRS);
    k_build_A<<<(BN * 16384) / 256, 256, 0, stream>>>(adjb, (float*)d_out);
}
